// MyLSTM_85985245266074
// MI455X (gfx1250) — compile-verified, matrix operand order checked by reading, not run on hardware
//
#include <hip/hip_runtime.h>
#include <math.h>

// ---------------------------------------------------------------------------
// CDNA5 (gfx1250) hierarchical LSTM tagger.
// All GEMMs use v_wmma_f32_16x16x32_f16 (wave32 WMMA). Operands are pre-packed
// into the ISA fragment layout so each lane loads one contiguous 32-byte v16h.
// Fragment layout (16-bit A 16x32 and B 32x16, wave32):
//   lane l: row/col = l & 15, khalf = l >> 4
//   half j (j<8):  K = kstep*32 + khalf*8 + j
//   half j (j>=8): K = kstep*32 + 16 + khalf*8 + (j-8)
// C/D f32 16x16: lane l: N = l&15 ; VGPR r: M = r + 8*(l>>4)
// Recurrence kernels keep h in LDS in *fragment order* (f16) so the per-step
// A operand is two ds_load_b128 per fragment instead of 16 scalar reads+cvt.
// ---------------------------------------------------------------------------

typedef __attribute__((ext_vector_type(16))) _Float16 v16h;
typedef __attribute__((ext_vector_type(8)))  float    v8f;

#define WMMA_F16(a, b, c) \
  __builtin_amdgcn_wmma_f32_16x16x32_f16(false, (a), false, (b), (short)0, (c), false, false)

__device__ __forceinline__ float sigmoidf_(float x) { return 1.0f / (1.0f + __expf(-x)); }
__device__ __forceinline__ float tanhf_fast(float x) { return 2.0f / (1.0f + __expf(-2.0f * x)) - 1.0f; }

// h[row][j] -> fragment slot  (fragments stored [kstep][lane][16] halves)
__device__ __forceinline__ int frag_slot(int row, int j) {
  const int ks    = j >> 5;
  const int k32   = j & 31;
  const int khalf = (k32 >> 3) & 1;
  const int jp    = (k32 & 7) + ((k32 & 16) ? 8 : 0);
  return ks * 512 + (row + 16 * khalf) * 16 + jp;
}

// Model dims
#define NB    32      // batch
#define NS    64      // sentence length
#define NWRD  2048    // NB*NS words
#define WLENT 20
#define NBERT 178
#define LCH   50      // letter features
#define TSEQ  198     // WLENT + NBERT
#define HS    128     // char-LSTM hidden
#define HM    256     // sentence-LSTM hidden
#define DWR   128     // word-rep dim
#define DIN   306     // DWR + NBERT
#define NTAG  32

// ---------------------------------------------------------------------------
// Weight packer: W is [N][K] row-major (used as x @ W.T, so B[k][n] = W[n][k]).
// Output fragments: [ntile][kstep][lane][16 halves], block index = ntile*Ksteps+kstep.
// block = 512 threads (lane = tid>>4, j = tid&15)
// ---------------------------------------------------------------------------
__global__ void pack_b_kernel(const float* __restrict__ W, _Float16* __restrict__ out,
                              int K, int Ksteps) {
  const int lane = threadIdx.x >> 4;
  const int j    = threadIdx.x & 15;
  const int blk  = blockIdx.x;                 // ntile*Ksteps + kstep
  const int ntile = blk / Ksteps, kstep = blk % Ksteps;
  const int n = ntile * 16 + (lane & 15);
  const int khalf = lane >> 4;
  const int k = kstep * 32 + ((j < 8) ? (khalf * 8 + j) : (16 + khalf * 8 + (j - 8)));
  float v = (k < K) ? W[(size_t)n * K + k] : 0.0f;
  out[(size_t)blk * 512 + lane * 16 + j] = (_Float16)v;
}

// Generic A packer: A fp32 [M][lda]
__global__ void pack_a_kernel(const float* __restrict__ A, _Float16* __restrict__ out,
                              int K, int lda, int Ksteps) {
  const int lane = threadIdx.x >> 4;
  const int j    = threadIdx.x & 15;
  const int blk  = blockIdx.x;                 // mtile*Ksteps + kstep
  const int mtile = blk / Ksteps, kstep = blk % Ksteps;
  const int m = mtile * 16 + (lane & 15);
  const int khalf = lane >> 4;
  const int k = kstep * 32 + ((j < 8) ? (khalf * 8 + j) : (16 + khalf * 8 + (j - 8)));
  float v = (k < K) ? A[(size_t)m * lda + k] : 0.0f;
  out[(size_t)blk * 512 + lane * 16 + j] = (_Float16)v;
}

// Letter A packer: rows m = word*20 + t, gathered from X [NWRD][TSEQ][LCH]
__global__ void pack_a_letters_kernel(const float* __restrict__ X, _Float16* __restrict__ out) {
  const int lane = threadIdx.x >> 4;
  const int j    = threadIdx.x & 15;
  const int blk  = blockIdx.x;                 // mtile*2 + kstep  (Ksteps = 2)
  const int mtile = blk >> 1, kstep = blk & 1;
  const int m = mtile * 16 + (lane & 15);
  const int w = m / WLENT, t = m % WLENT;
  const int khalf = lane >> 4;
  const int k = kstep * 32 + ((j < 8) ? (khalf * 8 + j) : (16 + khalf * 8 + (j - 8)));
  float v = (k < LCH) ? X[(size_t)w * (TSEQ * LCH) + t * LCH + k] : 0.0f;
  out[(size_t)blk * 512 + lane * 16 + j] = (_Float16)v;
}

// ---------------------------------------------------------------------------
// Generic packed WMMA GEMM: C[m, colOff + n] = A@B + bias0 + bias1
// grid = (Mtiles, ceil(Ntiles/8)), block = 256 (8 waves, 1 ntile per wave)
// ---------------------------------------------------------------------------
__global__ void gemm_wmma_kernel(const _Float16* __restrict__ Apk,
                                 const _Float16* __restrict__ Bpk,
                                 const float* __restrict__ bias0,
                                 const float* __restrict__ bias1,
                                 float* __restrict__ C,
                                 int Ksteps, int Ntiles, int ldc, int colOff) {
  const int lane  = threadIdx.x & 31;
  const int wave  = threadIdx.x >> 5;
  const int mtile = blockIdx.x;
  const int ntile = blockIdx.y * 8 + wave;
  if (ntile >= Ntiles) return;                 // wave-uniform
  const _Float16* ap = Apk + ((size_t)mtile * Ksteps) * 512 + lane * 16;
  const _Float16* bp = Bpk + ((size_t)ntile * Ksteps) * 512 + lane * 16;
  v8f acc = {};
  for (int ks = 0; ks < Ksteps; ++ks) {
    v16h a = *(const v16h*)(ap + (size_t)ks * 512);
    v16h b = *(const v16h*)(bp + (size_t)ks * 512);
    acc = WMMA_F16(a, b, acc);
  }
  const int n = ntile * 16 + (lane & 15);
  float bs = 0.0f;
  if (bias0) bs += bias0[n];
  if (bias1) bs += bias1[n];
  const int rbase = mtile * 16 + 8 * (lane >> 4);
#pragma unroll
  for (int r = 0; r < 8; ++r)
    C[(size_t)(rbase + r) * ldc + colOff + n] = acc[r] + bs;
}

// BERT feature gather into sentence buffer cols [128, 306)
__global__ void bert_fill_kernel(const float* __restrict__ X, float* __restrict__ sent) {
  const int w = blockIdx.x;
  const int j = threadIdx.x;
  if (j < NBERT)
    sent[(size_t)w * DIN + DWR + j] = X[(size_t)w * (TSEQ * LCH) + (WLENT + j) * LCH];
}

// ---------------------------------------------------------------------------
// Char LSTM: persistent block per 16 words, 20 sequential steps.
// Whh fragments (512x128 f16) live in registers: 8 waves x 4 ntiles x 4 ksteps.
// h kept in LDS as f16 A-fragments. grid = 128, block = 256.
// LDS: shf 4KB + sc 8KB + sg 32KB = 44 KB.
// ---------------------------------------------------------------------------
__global__ void word_lstm_kernel(const float* __restrict__ xg,      // [40960][512]
                                 const _Float16* __restrict__ Bpk,  // Whh_w packed
                                 const float* __restrict__ h0,
                                 const float* __restrict__ c0,
                                 float* __restrict__ hout) {        // [2048][128]
  __shared__ __align__(32) _Float16 shf[4 * 512];   // h as A-fragments
  __shared__ float sc[16 * HS];
  __shared__ float sg[16 * (4 * HS)];
  const int lane  = threadIdx.x & 31;
  const int wave  = threadIdx.x >> 5;
  const int mtile = blockIdx.x;

  v16h bw[4][4];
#pragma unroll
  for (int i = 0; i < 4; ++i) {
    const int ntile = wave * 4 + i;
#pragma unroll
    for (int ks = 0; ks < 4; ++ks)
      bw[i][ks] = *(const v16h*)(Bpk + (((size_t)ntile * 4 + ks) * 32 + lane) * 16);
  }
  for (int e = threadIdx.x; e < 16 * HS; e += blockDim.x) {
    const int row = e >> 7, jj = e & 127;
    shf[frag_slot(row, jj)] = (_Float16)h0[(size_t)(mtile * 16 + row) * HS + jj];
    sc[e] = c0[(size_t)(mtile * 16 + row) * HS + jj];
  }
  __syncthreads();

  for (int t = 0; t < WLENT; ++t) {
    // prefetch next step's gate pre-activations (16 rows x 512 f32 = 256 lines)
    if (t + 1 < WLENT) {
      const int prow = threadIdx.x >> 4, pline = threadIdx.x & 15;
      __builtin_prefetch(
          xg + (((size_t)(mtile * 16 + prow)) * WLENT + (t + 1)) * 512 + pline * 32, 0, 3);
    }
    v16h afr[4];
#pragma unroll
    for (int ks = 0; ks < 4; ++ks)
      afr[ks] = *(const v16h*)(&shf[ks * 512 + lane * 16]);
#pragma unroll
    for (int i = 0; i < 4; ++i) {
      const int ntile = wave * 4 + i;
      const int col = ntile * 16 + (lane & 15);
      const int rb  = 8 * (lane >> 4);
      v8f acc;
#pragma unroll
      for (int r = 0; r < 8; ++r)
        acc[r] = xg[(((size_t)(mtile * 16 + rb + r)) * WLENT + t) * 512 + col];
#pragma unroll
      for (int ks = 0; ks < 4; ++ks)
        acc = WMMA_F16(afr[ks], bw[i][ks], acc);
#pragma unroll
      for (int r = 0; r < 8; ++r)
        sg[(rb + r) * 512 + col] = acc[r];
    }
    __syncthreads();
    for (int e = threadIdx.x; e < 16 * HS; e += blockDim.x) {
      const int row = e >> 7, jj = e & 127;
      float gi = sg[row * 512 + jj];
      float gf = sg[row * 512 + 128 + jj];
      float gg = sg[row * 512 + 256 + jj];
      float go = sg[row * 512 + 384 + jj];
      float cc = sigmoidf_(gf) * sc[e] + sigmoidf_(gi) * tanhf_fast(gg);
      float hh = sigmoidf_(go) * tanhf_fast(cc);
      sc[e] = cc;
      shf[frag_slot(row, jj)] = (_Float16)hh;
      if (t == WLENT - 1) hout[(size_t)(mtile * 16 + row) * HS + jj] = hh;
    }
    __syncthreads();
  }
}

// ---------------------------------------------------------------------------
// Sentence LSTM (fwd/bwd): 2 persistent blocks x 16 rows, 64 sequential steps.
// Whh (1024x256 f16 = 512 KB) streamed from L2 each step as packed fragments.
// h kept in LDS as f16 A-fragments. grid = 2, block = 512 (16 waves x 4 ntiles).
// LDS: shf 8KB + sc 16KB + sg f16 32KB = 56 KB.
// ---------------------------------------------------------------------------
__global__ void sent_lstm_kernel(const float* __restrict__ xg,      // [2048][1024]
                                 const _Float16* __restrict__ Bpk,  // Whh packed
                                 const float* __restrict__ h0,
                                 const float* __restrict__ c0,
                                 float* __restrict__ tagin,         // [2048][512]
                                 int dir) {
  __shared__ __align__(32) _Float16 shf[8 * 512];   // h as A-fragments
  __shared__ float    sc[16 * HM];
  __shared__ _Float16 sg[16 * (4 * HM)];
  const int lane  = threadIdx.x & 31;
  const int wave  = threadIdx.x >> 5;
  const int mtile = blockIdx.x;

  for (int e = threadIdx.x; e < 16 * HM; e += blockDim.x) {
    const int row = e >> 8, jj = e & 255;
    shf[frag_slot(row, jj)] = (_Float16)h0[(size_t)(mtile * 16 + row) * HM + jj];
    sc[e] = c0[(size_t)(mtile * 16 + row) * HM + jj];
  }
  __syncthreads();

  for (int it = 0; it < NS; ++it) {
    const int s = dir ? (NS - 1 - it) : it;
    // prefetch next step's gate pre-activations (16 rows x 1024 f32 = 512 lines)
    if (it + 1 < NS) {
      const int snext = dir ? (s - 1) : (s + 1);
      const int prow = threadIdx.x >> 5, pline = threadIdx.x & 31;
      __builtin_prefetch(
          xg + ((size_t)(mtile * 16 + prow) * NS + snext) * 1024 + pline * 32, 0, 3);
    }
    v16h afr[8];
#pragma unroll
    for (int ks = 0; ks < 8; ++ks)
      afr[ks] = *(const v16h*)(&shf[ks * 512 + lane * 16]);
#pragma unroll
    for (int i = 0; i < 4; ++i) {
      const int ntile = wave * 4 + i;
      const int col = ntile * 16 + (lane & 15);
      const int rb  = 8 * (lane >> 4);
      v8f acc;
#pragma unroll
      for (int r = 0; r < 8; ++r) {
        const int brow = mtile * 16 + rb + r;
        acc[r] = xg[((size_t)brow * NS + s) * 1024 + col];
      }
#pragma unroll
      for (int ks = 0; ks < 8; ++ks) {
        v16h b = *(const v16h*)(Bpk + (((size_t)ntile * 8 + ks) * 32 + lane) * 16);
        acc = WMMA_F16(afr[ks], b, acc);
      }
#pragma unroll
      for (int r = 0; r < 8; ++r)
        sg[(rb + r) * 1024 + col] = (_Float16)acc[r];
    }
    __syncthreads();
    for (int e = threadIdx.x; e < 16 * HM; e += blockDim.x) {
      const int row = e >> 8, jj = e & 255;
      float gi = (float)sg[row * 1024 + jj];
      float gf = (float)sg[row * 1024 + 256 + jj];
      float gg = (float)sg[row * 1024 + 512 + jj];
      float go = (float)sg[row * 1024 + 768 + jj];
      float cc = sigmoidf_(gf) * sc[e] + sigmoidf_(gi) * tanhf_fast(gg);
      float hh = sigmoidf_(go) * tanhf_fast(cc);
      sc[e] = cc;
      shf[frag_slot(row, jj)] = (_Float16)hh;
      const int brow = mtile * 16 + row;
      tagin[((size_t)brow * NS + s) * 512 + dir * HM + jj] = hh;
    }
    __syncthreads();
  }
}

// ---------------------------------------------------------------------------
// Tag GEMM (2048 x 32, K = 512) + fused softmax. grid = 128, block = 64 (2 waves)
// ---------------------------------------------------------------------------
__global__ void tag_softmax_kernel(const _Float16* __restrict__ Apk,
                                   const _Float16* __restrict__ Bpk,
                                   const float* __restrict__ bo,
                                   float* __restrict__ out) {
  __shared__ float st[16 * NTAG];
  const int lane  = threadIdx.x & 31;
  const int wave  = threadIdx.x >> 5;     // 0..1 -> ntile
  const int mtile = blockIdx.x;
  {
    v8f acc = {};
    for (int ks = 0; ks < 16; ++ks) {
      v16h a = *(const v16h*)(Apk + (((size_t)mtile * 16 + ks) * 32 + lane) * 16);
      v16h b = *(const v16h*)(Bpk + (((size_t)wave * 16 + ks) * 32 + lane) * 16);
      acc = WMMA_F16(a, b, acc);
    }
    const int n = wave * 16 + (lane & 15);
    const int rb = 8 * (lane >> 4);
    const float bb = bo[n];
#pragma unroll
    for (int r = 0; r < 8; ++r) st[(rb + r) * NTAG + n] = acc[r] + bb;
  }
  __syncthreads();
  if (threadIdx.x < 16) {
    const int row = threadIdx.x;
    float mx = -3.0e38f;
    for (int c2 = 0; c2 < NTAG; ++c2) mx = fmaxf(mx, st[row * NTAG + c2]);
    float ex[NTAG];
    float sum = 0.0f;
    for (int c2 = 0; c2 < NTAG; ++c2) { ex[c2] = __expf(st[row * NTAG + c2] - mx); sum += ex[c2]; }
    const float inv = 1.0f / sum;
    for (int c2 = 0; c2 < NTAG; ++c2)
      out[(size_t)(mtile * 16 + row) * NTAG + c2] = ex[c2] * inv;
  }
}

// ---------------------------------------------------------------------------
extern "C" void kernel_launch(void* const* d_in, const int* in_sizes, int n_in,
                              void* d_out, int out_size, void* d_ws, size_t ws_size,
                              hipStream_t stream) {
  const float* X     = (const float*)d_in[0];
  const float* h0w   = (const float*)d_in[1];
  const float* c0w   = (const float*)d_in[2];
  const float* h0f   = (const float*)d_in[3];
  const float* c0f   = (const float*)d_in[4];
  const float* h0b   = (const float*)d_in[5];
  const float* c0b   = (const float*)d_in[6];
  const float* Wih_w = (const float*)d_in[7];
  const float* Whh_w = (const float*)d_in[8];
  const float* bih_w = (const float*)d_in[9];
  const float* bhh_w = (const float*)d_in[10];
  const float* Wl    = (const float*)d_in[11];
  const float* bl    = (const float*)d_in[12];
  const float* Wih_f = (const float*)d_in[13];
  const float* Whh_f = (const float*)d_in[14];
  const float* bih_f = (const float*)d_in[15];
  const float* bhh_f = (const float*)d_in[16];
  const float* Wih_b = (const float*)d_in[17];
  const float* Whh_b = (const float*)d_in[18];
  const float* bih_b = (const float*)d_in[19];
  const float* bhh_b = (const float*)d_in[20];
  const float* Wo    = (const float*)d_in[21];
  const float* bo    = (const float*)d_in[22];

  char* ws = (char*)d_ws;
  size_t off = 0;
  auto take = [&](size_t bytes) -> void* {
    void* p = ws + off;
    off += (bytes + 255) & ~(size_t)255;
    return p;
  };
  _Float16* pIhW  = (_Float16*)take((size_t)32 * 2 * 512 * 2);   // 512x64
  _Float16* pHhW  = (_Float16*)take((size_t)32 * 4 * 512 * 2);   // 512x128
  _Float16* pWl   = (_Float16*)take((size_t)8  * 4 * 512 * 2);   // 128x128
  _Float16* pIhF  = (_Float16*)take((size_t)64 * 10 * 512 * 2);  // 1024x320
  _Float16* pHhF  = (_Float16*)take((size_t)64 * 8  * 512 * 2);  // 1024x256
  _Float16* pIhB  = (_Float16*)take((size_t)64 * 10 * 512 * 2);
  _Float16* pHhB  = (_Float16*)take((size_t)64 * 8  * 512 * 2);
  _Float16* pWo   = (_Float16*)take((size_t)2  * 16 * 512 * 2);  // 32x512
  _Float16* aLet  = (_Float16*)take((size_t)2560 * 2 * 512 * 2);
  float*    xgw   = (float*)   take((size_t)40960 * 512 * 4);
  float*    hwfin = (float*)   take((size_t)NWRD * HS * 4);
  _Float16* aHw   = (_Float16*)take((size_t)128 * 4 * 512 * 2);
  float*    sent  = (float*)   take((size_t)NWRD * DIN * 4);
  _Float16* aSent = (_Float16*)take((size_t)128 * 10 * 512 * 2);
  float*    xgf   = (float*)   take((size_t)NWRD * 1024 * 4);
  float*    xgb   = (float*)   take((size_t)NWRD * 1024 * 4);
  float*    tagin = (float*)   take((size_t)NWRD * 512 * 4);
  _Float16* aTag  = (_Float16*)take((size_t)128 * 16 * 512 * 2);
  (void)ws_size; (void)in_sizes; (void)n_in; (void)out_size;

  // 1) pack all weights into WMMA B fragments (f16)
  pack_b_kernel<<<32 * 2,  512, 0, stream>>>(Wih_w, pIhW, LCH, 2);
  pack_b_kernel<<<32 * 4,  512, 0, stream>>>(Whh_w, pHhW, HS, 4);
  pack_b_kernel<<<8  * 4,  512, 0, stream>>>(Wl,    pWl,  HS, 4);
  pack_b_kernel<<<64 * 10, 512, 0, stream>>>(Wih_f, pIhF, DIN, 10);
  pack_b_kernel<<<64 * 8,  512, 0, stream>>>(Whh_f, pHhF, HM, 8);
  pack_b_kernel<<<64 * 10, 512, 0, stream>>>(Wih_b, pIhB, DIN, 10);
  pack_b_kernel<<<64 * 8,  512, 0, stream>>>(Whh_b, pHhB, HM, 8);
  pack_b_kernel<<<2  * 16, 512, 0, stream>>>(Wo,    pWo,  512, 16);

  // 2) char-LSTM input projection: xg_w = letters @ Wih_w.T + bih_w + bhh_w
  pack_a_letters_kernel<<<2560 * 2, 512, 0, stream>>>(X, aLet);
  gemm_wmma_kernel<<<dim3(2560, 4), 256, 0, stream>>>(aLet, pIhW, bih_w, bhh_w,
                                                      xgw, 2, 32, 512, 0);

  // 3) char-LSTM recurrence (persistent, Whh in registers)
  word_lstm_kernel<<<128, 256, 0, stream>>>(xgw, pHhW, h0w, c0w, hwfin);

  // 4) word_rep = h_final @ Wl.T + bl  -> sentence buffer cols [0,128)
  pack_a_kernel<<<128 * 4, 512, 0, stream>>>(hwfin, aHw, HS, HS, 4);
  gemm_wmma_kernel<<<dim3(128, 1), 256, 0, stream>>>(aHw, pWl, bl, nullptr,
                                                     sent, 4, 8, DIN, 0);
  bert_fill_kernel<<<NWRD, 192, 0, stream>>>(X, sent);

  // 5) sentence-LSTM input projections
  pack_a_kernel<<<128 * 10, 512, 0, stream>>>(sent, aSent, DIN, DIN, 10);
  gemm_wmma_kernel<<<dim3(128, 8), 256, 0, stream>>>(aSent, pIhF, bih_f, bhh_f,
                                                     xgf, 10, 64, 1024, 0);
  gemm_wmma_kernel<<<dim3(128, 8), 256, 0, stream>>>(aSent, pIhB, bih_b, bhh_b,
                                                     xgb, 10, 64, 1024, 0);

  // 6) bidirectional sentence-LSTM recurrences -> concat buffer [2048][512]
  sent_lstm_kernel<<<2, 512, 0, stream>>>(xgf, pHhF, h0f, c0f, tagin, 0);
  sent_lstm_kernel<<<2, 512, 0, stream>>>(xgb, pHhB, h0b, c0b, tagin, 1);

  // 7) tags = tagin @ Wo.T + bo ; softmax
  pack_a_kernel<<<128 * 16, 512, 0, stream>>>(tagin, aTag, 512, 512, 16);
  tag_softmax_kernel<<<128, 64, 0, stream>>>(aTag, pWo, bo, (float*)d_out);
}
